// MixedDiffEncoder_67551245631815
// MI455X (gfx1250) — matrix operand order repaired and checked by reading, not compiled, not run on hardware
//
#include <hip/hip_runtime.h>
#include <hip/hip_bf16.h>
#include <math.h>
#include <stdint.h>

#define B_   64
#define D_   3072
#define K_   512
#define E2_  64
#define BETA_ 5.0f

typedef __attribute__((ext_vector_type(16))) _Float16 v16h;
typedef __attribute__((ext_vector_type(8)))  float    v8f;
typedef unsigned int u32x4 __attribute__((ext_vector_type(4)));
typedef int          i32x4 __attribute__((ext_vector_type(4)));
typedef int          i32x8 __attribute__((ext_vector_type(8)));

union HF { v16h v; _Float16 h[16]; };

// A-fragment (16-bit, 16x32) K index for element slot idx (0..15) given lane group (lane>>4).
// Contiguous order matching B: lanes 0-15 hold K=0..15; lanes 16-31 hold K=16..31.
__device__ __forceinline__ int a_kidx(int grp, int idx) {
  return grp * 16 + idx;
}

__device__ __forceinline__ v8f wmma16(v16h a, v16h b, v8f c) {
  return __builtin_amdgcn_wmma_f32_16x16x32_f16(false, a, false, b, (short)0, c, false, false);
}

// ---- Tensor Data Mover: 1D contiguous tile (n_dwords x 4B) global -> LDS ----
// D# per ISA 8.3/8.4: group0 = {count=1 | lds_addr | global_addr | type=2},
// group1 = {data_size=4B, tensor_dim0=n, tensor_dim1=1, tile_dim0=n, tile_dim1=1,
//           tensor_dim0_stride=n}. Groups 2/3 zero (<=2D tensor).
__device__ __forceinline__ void tdm_load_tile_f32(void* lds_dst, const float* gsrc,
                                                  unsigned n_dwords) {
  uint64_t ga = (uint64_t)(uintptr_t)gsrc;
  uint32_t lo = (uint32_t)(uintptr_t)lds_dst;  // flat addr low 32 bits == LDS byte address
  u32x4 g0 = {0u, 0u, 0u, 0u};
  i32x8 g1 = {0, 0, 0, 0, 0, 0, 0, 0};
  i32x4 gz = {0, 0, 0, 0};
  g0[0] = 1u;                                           // count=1 (valid user D#)
  g0[1] = lo;                                           // lds_addr  [63:32]
  g0[2] = (uint32_t)ga;                                 // global_addr[31:0]
  g0[3] = (uint32_t)((ga >> 32) & 0x01FFFFFFu) | (2u << 30);  // [56:32] | type=2
  g1[0] = (int)(2u << 16);                              // data_size = 2 (4 bytes)
  g1[1] = (int)((n_dwords & 0xFFFFu) << 16);            // tensor_dim0[15:0]  @bit48
  g1[2] = (int)(((n_dwords >> 16) & 0xFFFFu) | (1u << 16)); // tensor_dim0[31:16], tensor_dim1=1
  g1[3] = (int)((n_dwords & 0xFFFFu) << 16);            // tile_dim0 @bit112 (fits 16b)
  g1[4] = 1;                                            // tile_dim1 = 1, tile_dim2 = 0
  g1[5] = (int)n_dwords;                                // tensor_dim0_stride[31:0]
#if defined(__clang_major__) && (__clang_major__ >= 23)
  i32x8 gz8 = {0, 0, 0, 0, 0, 0, 0, 0};
  __builtin_amdgcn_tensor_load_to_lds(g0, g1, gz, gz, gz8, 0);
#else
  __builtin_amdgcn_tensor_load_to_lds(g0, g1, gz, gz, 0);
#endif
}

// ---------------------------------------------------------------- prep -----
__global__ void k_prep_convert(const float* __restrict__ imgs, const float* __restrict__ wp,
                               _Float16* __restrict__ imgsh, _Float16* __restrict__ wph,
                               float* __restrict__ xd2) {
  int stride = gridDim.x * blockDim.x;
  int tid = blockIdx.x * blockDim.x + threadIdx.x;
  for (int i = tid; i < B_ * D_; i += stride) imgsh[i] = (_Float16)imgs[i];
  for (int i = tid; i < K_ * D_; i += stride) wph[i] = (_Float16)wp[i];
  for (int i = tid; i < B_ * D_; i += stride) xd2[i] = 0.f;
}

__global__ void k_prep_norms(const float* __restrict__ imgs, const float* __restrict__ wp,
                             float* __restrict__ x2, float* __restrict__ w2,
                             float* __restrict__ out) {
  int t = threadIdx.x;
  if (blockIdx.x == 0) {
    __shared__ float red[256];
    int b = t & 63, q = t >> 6;
    float s = 0.f;
    for (int d = q * (D_ / 4); d < (q + 1) * (D_ / 4); ++d) {
      float v = imgs[(size_t)b * D_ + d];
      s += v * v;
    }
    red[t] = s;
    __syncthreads();
    if (t < 64) {
      x2[t] = red[t] + red[t + 64] + red[t + 128] + red[t + 192];
      out[t] = 0.f;   // zero loss accumulator every launch (graph replay safe)
    }
  } else {
    int k = (blockIdx.x - 1) * 256 + t;   // blocks 1,2 -> k in [0,512)
    float s = 0.f;
    for (int d = 0; d < D_; ++d) {
      float v = wp[(size_t)k * D_ + d];
      s += v * v;
    }
    w2[k] = s;
  }
}

// ------------------------------------------- G = images @ w_proj^T (WMMA) --
__global__ void k_proj_gemm(const _Float16* __restrict__ imgsh, const _Float16* __restrict__ wph,
                            float* __restrict__ G) {
  __shared__ __align__(16) _Float16 shA[64][72];   // [b][d]
  __shared__ __align__(16) _Float16 shB[64][72];   // [kcol][d]
  int t = threadIdx.x, lane = t & 31, wid = t >> 5, grp = lane >> 4;
  int k0 = blockIdx.x * 64;
  int ti0 = wid * 2, ti1 = ti0 + 1;
  int mi0 = ti0 >> 2, ni0 = ti0 & 3, mi1 = ti1 >> 2, ni1 = ti1 & 3;
  v8f acc0 = {}, acc1 = {};
  for (int d0 = 0; d0 < D_; d0 += 64) {
    for (int c = t; c < 512; c += 256) {
      int r = c >> 3, dc = (c & 7) * 8;
      *(float4*)&shA[r][dc] = *(const float4*)&imgsh[(size_t)r * D_ + d0 + dc];
      *(float4*)&shB[r][dc] = *(const float4*)&wph[(size_t)(k0 + r) * D_ + d0 + dc];
    }
    __syncthreads();
#pragma unroll
    for (int kk = 0; kk < 2; ++kk) {
      HF a0, b0, a1, b1;
#pragma unroll
      for (int idx = 0; idx < 16; ++idx) {
        int ak = kk * 32 + a_kidx(grp, idx);
        int bk = kk * 32 + grp * 16 + idx;
        a0.h[idx] = shA[mi0 * 16 + (lane & 15)][ak];
        a1.h[idx] = shA[mi1 * 16 + (lane & 15)][ak];
        b0.h[idx] = shB[ni0 * 16 + (lane & 15)][bk];
        b1.h[idx] = shB[ni1 * 16 + (lane & 15)][bk];
      }
      acc0 = wmma16(a0.v, b0.v, acc0);
      acc1 = wmma16(a1.v, b1.v, acc1);
    }
    __syncthreads();
  }
#pragma unroll
  for (int r = 0; r < 8; ++r) {
    int M0 = mi0 * 16 + grp * 8 + r, N0 = ni0 * 16 + (lane & 15);
    int M1 = mi1 * 16 + grp * 8 + r, N1 = ni1 * 16 + (lane & 15);
    G[(size_t)M0 * K_ + k0 + N0] = acc0[r];
    G[(size_t)M1 * K_ + k0 + N1] = acc1[r];
  }
}

// -------- big encode pass: A[k,b,e]=img@raw[k]; fused stats -> s,c (WMMA) --
// Raw f32 tiles streamed LDS-ward by the Tensor Data Mover, double-buffered.
__global__ void k_encode_gemm(const float* __restrict__ wimg2, const float* __restrict__ wp,
                              const _Float16* __restrict__ imgsh,
                              float* __restrict__ Aout, float* __restrict__ sArr,
                              float* __restrict__ cArr) {
  __shared__ __align__(16) float bufF[2][64 * 64];  // TDM landing buffers (f32 tiles)
  __shared__ __align__(16) _Float16 shA[64][72];    // images [b][d]
  __shared__ __align__(16) _Float16 shR[64][72];    // raw    [d][e] (f16)
  __shared__ float shWp[64];
  __shared__ float red[256];
  int t = threadIdx.x, lane = t & 31, wid = t >> 5, grp = lane >> 4;
  int k = blockIdx.x;
  int e = t & 63, dq = t >> 6;
  int ti0 = wid * 2, ti1 = ti0 + 1;
  int mi0 = ti0 >> 2, ni0 = ti0 & 3, mi1 = ti1 >> 2, ni1 = ti1 & 3;
  float cp = 0.f, m1p = 0.f, m2p = 0.f;
  v8f acc0 = {}, acc1 = {};
  const float* rawk = wimg2 + (size_t)k * D_ * E2_;
  const int NT = D_ / 64;   // 48 tiles
  if (t < 32) tdm_load_tile_f32(&bufF[0][0], rawk, 64 * 64);        // prime tile 0
  for (int dt = 0; dt < NT; ++dt) {
    int cur = dt & 1;
    if (t < 32) {
      if (dt + 1 < NT) {
        tdm_load_tile_f32(&bufF[1 - cur][0], rawk + (size_t)(dt + 1) * 64 * E2_, 64 * 64);
        __builtin_amdgcn_s_wait_tensorcnt(1);   // all but the just-issued complete
      } else {
        __builtin_amdgcn_s_wait_tensorcnt(0);   // drain on last tile
      }
    }
    for (int c = t; c < 512; c += 256) {
      int r = c >> 3, dc = (c & 7) * 8;
      *(float4*)&shA[r][dc] = *(const float4*)&imgsh[(size_t)r * D_ + dt * 64 + dc];
    }
    if (t < 64) shWp[t] = wp[(size_t)k * D_ + dt * 64 + t];
    __syncthreads();                                  // tile dt visible to all waves
#pragma unroll 4
    for (int i = 0; i < 16; ++i) {                    // stats + f32 -> f16 convert
      int d = dq * 16 + i;
      float v = bufF[cur][d * 64 + e];
      cp += shWp[d] * v; m1p += v; m2p += v * v;
      shR[d][e] = (_Float16)v;
    }
    __syncthreads();
#pragma unroll
    for (int kk = 0; kk < 2; ++kk) {
      HF a0, b0, a1, b1;
#pragma unroll
      for (int idx = 0; idx < 16; ++idx) {
        int ak = kk * 32 + a_kidx(grp, idx);
        int bk = kk * 32 + grp * 16 + idx;
        a0.h[idx] = shA[mi0 * 16 + (lane & 15)][ak];
        a1.h[idx] = shA[mi1 * 16 + (lane & 15)][ak];
        b0.h[idx] = shR[bk][ni0 * 16 + (lane & 15)];
        b1.h[idx] = shR[bk][ni1 * 16 + (lane & 15)];
      }
      acc0 = wmma16(a0.v, b0.v, acc0);
      acc1 = wmma16(a1.v, b1.v, acc1);
    }
    __syncthreads();
  }
  // reduce per-e stats over the 4 dq groups
  red[t] = cp; __syncthreads();
  float cs = 0.f;
  if (t < 64) cs = red[t] + red[t + 64] + red[t + 128] + red[t + 192];
  __syncthreads(); red[t] = m1p; __syncthreads();
  float m1s = 0.f;
  if (t < 64) m1s = red[t] + red[t + 64] + red[t + 128] + red[t + 192];
  __syncthreads(); red[t] = m2p; __syncthreads();
  if (t < 64) {
    float m2s = red[t] + red[t + 64] + red[t + 128] + red[t + 192];
    float mean = m1s / (float)D_;
    float var = (m2s - (float)D_ * mean * mean) / (float)(D_ - 1);  // ddof=1
    sArr[k * E2_ + t] = 1.f / (0.01f + sqrtf(fmaxf(var, 0.f)));
    cArr[k * E2_ + t] = cs;
  }
#pragma unroll
  for (int r = 0; r < 8; ++r) {
    int M0 = mi0 * 16 + grp * 8 + r, N0 = ni0 * 16 + (lane & 15);
    int M1 = mi1 * 16 + grp * 8 + r, N1 = ni1 * 16 + (lane & 15);
    Aout[((size_t)k * B_ + M0) * E2_ + N0] = acc0[r];
    Aout[((size_t)k * B_ + M1) * E2_ + N1] = acc1[r];
  }
}

// ------------------------------------ encode softmax, xp, lat1 = xp@wrec ---
__global__ void k_encode_softmax(const float* __restrict__ G, const float* __restrict__ x2,
                                 const float* __restrict__ w2, const float* __restrict__ w_rec,
                                 float* __restrict__ xp, float* __restrict__ lat1) {
  __shared__ float shXp[512];
  __shared__ float red[256];
  int b = blockIdx.x, t = threadIdx.x;
  float x2b = x2[b];
  float h0 = -(BETA_ / (float)D_) * (x2b - 2.f * G[(size_t)b * K_ + t]       + w2[t]);
  float h1 = -(BETA_ / (float)D_) * (x2b - 2.f * G[(size_t)b * K_ + t + 256] + w2[t + 256]);
  red[t] = fmaxf(h0, h1); __syncthreads();
  for (int s = 128; s > 0; s >>= 1) { if (t < s) red[t] = fmaxf(red[t], red[t + s]); __syncthreads(); }
  float mx = red[0]; __syncthreads();
  float e0 = __expf(h0 - mx), e1 = __expf(h1 - mx);
  red[t] = e0 + e1; __syncthreads();
  for (int s = 128; s > 0; s >>= 1) { if (t < s) red[t] += red[t + s]; __syncthreads(); }
  float inv = 1.f / red[0]; __syncthreads();
  shXp[t] = e0 * inv; shXp[t + 256] = e1 * inv;
  xp[(size_t)b * K_ + t] = e0 * inv; xp[(size_t)b * K_ + t + 256] = e1 * inv;
  __syncthreads();
  int e = t & 63, kq = t >> 6;
  float acc = 0.f;
  for (int k = kq; k < K_; k += 4) acc += shXp[k] * w_rec[(size_t)e * K_ + k];
  red[t] = acc; __syncthreads();
  if (t < 64) lat1[b * E2_ + t] = red[t] + red[t + 64] + red[t + 128] + red[t + 192];
}

// -------------------------- lat2[b,e] = sum_k xp * s*(A - c) ---------------
__global__ void k_reduce_lat2(const float* __restrict__ A, const float* __restrict__ sArr,
                              const float* __restrict__ cArr, const float* __restrict__ xp,
                              float* __restrict__ lat2) {
  __shared__ float red[256];
  int b = blockIdx.x, t = threadIdx.x, e = t & 63, kq = t >> 6;
  float acc = 0.f;
  for (int k = kq; k < K_; k += 4) {
    float xe = sArr[k * E2_ + e] * (A[((size_t)k * B_ + b) * E2_ + e] - cArr[k * E2_ + e]);
    acc += xp[(size_t)b * K_ + k] * xe;
  }
  red[t] = acc; __syncthreads();
  if (t < 64) lat2[b * E2_ + t] = red[t] + red[t + 64] + red[t + 128] + red[t + 192];
}

// --------------------------------------- decode softmax -> xp2 -------------
__global__ void k_decode_softmax(const float* __restrict__ lat1, const float* __restrict__ w_rec,
                                 float* __restrict__ xp2) {
  __shared__ float shL[64];
  __shared__ float red[256];
  int b = blockIdx.x, t = threadIdx.x;
  if (t < 64) shL[t] = lat1[b * E2_ + t];
  __syncthreads();
  float d0s = 0.f, d1s = 0.f;
  for (int e = 0; e < E2_; ++e) {
    float l = shL[e];
    float a = l - w_rec[(size_t)e * K_ + t];        d0s += a * a;
    float c = l - w_rec[(size_t)e * K_ + t + 256];  d1s += c * c;
  }
  float h0 = -(BETA_ / (float)E2_) * d0s, h1 = -(BETA_ / (float)E2_) * d1s;
  red[t] = fmaxf(h0, h1); __syncthreads();
  for (int s = 128; s > 0; s >>= 1) { if (t < s) red[t] = fmaxf(red[t], red[t + s]); __syncthreads(); }
  float mx = red[0]; __syncthreads();
  float e0 = __expf(h0 - mx), e1 = __expf(h1 - mx);
  red[t] = e0 + e1; __syncthreads();
  for (int s = 128; s > 0; s >>= 1) { if (t < s) red[t] += red[t + s]; __syncthreads(); }
  float inv = 1.f / red[0];
  xp2[(size_t)b * K_ + t] = e0 * inv; xp2[(size_t)b * K_ + t + 256] = e1 * inv;
}

// -------- big decode pass: xd2 += (xp2*lat2*s) x raw  (WMMA + TDM) ---------
__global__ void k_decode_big(const float* __restrict__ wimg2, const float* __restrict__ sArr,
                             const float* __restrict__ xp2, const float* __restrict__ lat2,
                             float* __restrict__ xd2) {
  __shared__ __align__(16) float bufF[2][64 * 64];  // TDM landing buffers
  __shared__ __align__(16) _Float16 shR[64][72];    // raw [d][e] (f16)
  __shared__ float shL2[64][65];                    // lat2 [b][e]
  __shared__ float shS[64];
  __shared__ float shX[64];
  int t = threadIdx.x, lane = t & 31, wid = t >> 5, grp = lane >> 4;
  int d0 = blockIdx.x * 64;
  int kbase = blockIdx.y * 64;
  int e = t & 63, dq = t >> 6;
  int ti0 = wid * 2, ti1 = ti0 + 1;
  int mi0 = ti0 >> 2, ni0 = ti0 & 3, mi1 = ti1 >> 2, ni1 = ti1 & 3;
  for (int i = t; i < B_ * E2_; i += 256) shL2[i >> 6][i & 63] = lat2[i];
  v8f acc0 = {}, acc1 = {};
  if (t < 32)
    tdm_load_tile_f32(&bufF[0][0], wimg2 + ((size_t)kbase * D_ + d0) * E2_, 64 * 64);
  for (int kv = 0; kv < 64; ++kv) {
    int k = kbase + kv;
    int cur = kv & 1;
    if (t < 32) {
      if (kv + 1 < 64) {
        tdm_load_tile_f32(&bufF[1 - cur][0],
                          wimg2 + ((size_t)(k + 1) * D_ + d0) * E2_, 64 * 64);
        __builtin_amdgcn_s_wait_tensorcnt(1);
      } else {
        __builtin_amdgcn_s_wait_tensorcnt(0);
      }
    }
    if (t < 64) { shS[t] = sArr[k * E2_ + t]; shX[t] = xp2[(size_t)t * K_ + k]; }
    __syncthreads();                                  // tile kv visible to all waves
#pragma unroll 4
    for (int i = 0; i < 16; ++i) {
      int d = dq * 16 + i;
      shR[d][e] = (_Float16)bufF[cur][d * 64 + e];
    }
    __syncthreads();
#pragma unroll
    for (int kk = 0; kk < 2; ++kk) {
      HF a0, b0, a1, b1;
      int bb0 = mi0 * 16 + (lane & 15), bb1 = mi1 * 16 + (lane & 15);
#pragma unroll
      for (int idx = 0; idx < 16; ++idx) {
        int ea = kk * 32 + a_kidx(grp, idx);          // inner dim = e
        int eb = kk * 32 + grp * 16 + idx;
        a0.h[idx] = (_Float16)(shX[bb0] * shL2[bb0][ea] * shS[ea]);
        a1.h[idx] = (_Float16)(shX[bb1] * shL2[bb1][ea] * shS[ea]);
        b0.h[idx] = shR[ni0 * 16 + (lane & 15)][eb];  // N = d column
        b1.h[idx] = shR[ni1 * 16 + (lane & 15)][eb];
      }
      acc0 = wmma16(a0.v, b0.v, acc0);
      acc1 = wmma16(a1.v, b1.v, acc1);
    }
    __syncthreads();
  }
#pragma unroll
  for (int r = 0; r < 8; ++r) {
    int M0 = mi0 * 16 + grp * 8 + r, N0 = ni0 * 16 + (lane & 15);
    int M1 = mi1 * 16 + grp * 8 + r, N1 = ni1 * 16 + (lane & 15);
    atomicAdd(&xd2[(size_t)M0 * D_ + d0 + N0], acc0[r]);
    atomicAdd(&xd2[(size_t)M1 * D_ + d0 + N1], acc1[r]);
  }
}

// -------- recon = xp2 @ w_proj + xd2 ; loss[b] = mean_d (recon-img)^2 ------
__global__ void k_decode_recon(const _Float16* __restrict__ wph, const float* __restrict__ xp2,
                               const float* __restrict__ xd2, const float* __restrict__ imgs,
                               float* __restrict__ out) {
  __shared__ __align__(16) _Float16 shW[64][72];  // [kc][d]
  __shared__ float shXp[64][65];                  // [b][kc]
  __shared__ float shLoss[64];
  int t = threadIdx.x, lane = t & 31, wid = t >> 5, grp = lane >> 4;
  int d0 = blockIdx.x * 64;
  int ti0 = wid * 2, ti1 = ti0 + 1;
  int mi0 = ti0 >> 2, ni0 = ti0 & 3, mi1 = ti1 >> 2, ni1 = ti1 & 3;
  v8f acc0, acc1;
#pragma unroll
  for (int r = 0; r < 8; ++r) {   // init accumulators from xd2
    acc0[r] = xd2[(size_t)(mi0 * 16 + grp * 8 + r) * D_ + d0 + ni0 * 16 + (lane & 15)];
    acc1[r] = xd2[(size_t)(mi1 * 16 + grp * 8 + r) * D_ + d0 + ni1 * 16 + (lane & 15)];
  }
  if (t < 64) shLoss[t] = 0.f;
  for (int kt = 0; kt < 8; ++kt) {
    int kb = kt * 64;
    for (int c = t; c < 512; c += 256) {
      int r = c >> 3, dc = (c & 7) * 8;
      *(float4*)&shW[r][dc] = *(const float4*)&wph[(size_t)(kb + r) * D_ + d0 + dc];
    }
    for (int i = 0; i < 16; ++i) {
      int bb = (t >> 6) * 16 + i;
      shXp[bb][t & 63] = xp2[(size_t)bb * K_ + kb + (t & 63)];
    }
    __syncthreads();
#pragma unroll
    for (int kk = 0; kk < 2; ++kk) {
      HF a0, b0, a1, b1;
#pragma unroll
      for (int idx = 0; idx < 16; ++idx) {
        int ak = kk * 32 + a_kidx(grp, idx);
        int bk = kk * 32 + grp * 16 + idx;
        a0.h[idx] = (_Float16)shXp[mi0 * 16 + (lane & 15)][ak];
        a1.h[idx] = (_Float16)shXp[mi1 * 16 + (lane & 15)][ak];
        b0.h[idx] = shW[bk][ni0 * 16 + (lane & 15)];
        b1.h[idx] = shW[bk][ni1 * 16 + (lane & 15)];
      }
      acc0 = wmma16(a0.v, b0.v, acc0);
      acc1 = wmma16(a1.v, b1.v, acc1);
    }
    __syncthreads();
  }
#pragma unroll
  for (int r = 0; r < 8; ++r) {
    int M0 = mi0 * 16 + grp * 8 + r, N0 = d0 + ni0 * 16 + (lane & 15);
    int M1 = mi1 * 16 + grp * 8 + r, N1 = d0 + ni1 * 16 + (lane & 15);
    float f0 = acc0[r] - imgs[(size_t)M0 * D_ + N0];
    float f1 = acc1[r] - imgs[(size_t)M1 * D_ + N1];
    atomicAdd(&shLoss[M0], f0 * f0);
    atomicAdd(&shLoss[M1], f1 * f1);
  }
  __syncthreads();
  if (t < 64) atomicAdd(&out[t], shLoss[t] * (1.0f / (float)D_));
}

// ---------------------------------------------------------------------------
extern "C" void kernel_launch(void* const* d_in, const int* in_sizes, int n_in,
                              void* d_out, int out_size, void* d_ws, size_t ws_size,
                              hipStream_t stream) {
  (void)in_sizes; (void)n_in; (void)out_size; (void)ws_size;
  const float* imgs  = (const float*)d_in[0];
  const float* wproj = (const float*)d_in[1];
  const float* wimg2 = (const float*)d_in[2];
  const float* wrec  = (const float*)d_in[3];
  float* out = (float*)d_out;
  char* ws = (char*)d_ws;
  // workspace layout (bytes), total ~13.4 MB
  _Float16* imgsh = (_Float16*)(ws + 0);          // 393216
  _Float16* wph   = (_Float16*)(ws + 393216);     // 3145728
  float* A    = (float*)(ws + 3538944);           // 8388608
  float* sArr = (float*)(ws + 11927552);          // 131072
  float* cArr = (float*)(ws + 12058624);          // 131072
  float* G    = (float*)(ws + 12189696);          // 131072
  float* xp   = (float*)(ws + 12320768);          // 131072
  float* xp2  = (float*)(ws + 12451840);          // 131072
  float* lat1 = (float*)(ws + 12582912);          // 16384
  float* lat2 = (float*)(ws + 12599296);          // 16384
  float* w2   = (float*)(ws + 12615680);          // 2048
  float* x2   = (float*)(ws + 12617728);          // 256
  float* xd2  = (float*)(ws + 12617984);          // 786432

  k_prep_convert<<<432, 256, 0, stream>>>(imgs, wproj, imgsh, wph, xd2);
  k_prep_norms<<<3, 256, 0, stream>>>(imgs, wproj, x2, w2, out);
  k_proj_gemm<<<8, 256, 0, stream>>>(imgsh, wph, G);
  k_encode_gemm<<<512, 256, 0, stream>>>(wimg2, wproj, imgsh, A, sArr, cArr);
  k_encode_softmax<<<64, 256, 0, stream>>>(G, x2, w2, wrec, xp, lat1);
  k_reduce_lat2<<<64, 256, 0, stream>>>(A, sArr, cArr, xp, lat2);
  k_decode_softmax<<<64, 256, 0, stream>>>(lat1, wrec, xp2);
  k_decode_big<<<dim3(48, 8), 256, 0, stream>>>(wimg2, sArr, xp2, lat2, xd2);
  k_decode_recon<<<48, 256, 0, stream>>>(wph, xp2, xd2, imgs, out);
}
